// TransformerLM_7679401525269
// MI455X (gfx1250) — matrix operand order repaired and checked by reading, not compiled, not run on hardware
//
#include <hip/hip_runtime.h>
#include <hip/hip_bf16.h>

// ---------------------------------------------------------------------------
// CDNA5 (gfx1250) transformer forward. GEMMs via v_wmma_f32_16x16x32_bf16
// (fp32 accumulate, wave32). A-tiles staged by the Tensor Data Mover
// (tensor_load_to_lds + s_wait_tensorcnt), attention K^T fragments via
// global_load_b128 row loads (no transpose). Layouts per CDNA5 ISA 7.12.2.
// ---------------------------------------------------------------------------

typedef __attribute__((ext_vector_type(16))) __bf16       v16bf;
typedef __attribute__((ext_vector_type(8)))  __bf16       v8bf;
typedef __attribute__((ext_vector_type(8)))  float        v8f;
typedef __attribute__((ext_vector_type(4)))  unsigned int u32x4;
typedef __attribute__((ext_vector_type(8)))  int          i32x8;
typedef __attribute__((ext_vector_type(4)))  int          i32x4;

#define WMMA_BF16(a, b, c) \
  __builtin_amdgcn_wmma_f32_16x16x32_bf16(false, (a), false, (b), (short)0, (c), false, false)

#if defined(__has_builtin)
#if __has_builtin(__builtin_amdgcn_tensor_load_to_lds) && \
    __has_builtin(__builtin_amdgcn_s_wait_tensorcnt)
#define USE_TDM 1
#endif
#endif

constexpr int BB  = 2;
constexpr int SS  = 1024;
constexpr int DD  = 1024;
constexpr int HHN = 16;
constexpr int DHD = 64;
constexpr int FF  = 4096;
constexpr int VV  = 32000;
constexpr int LLN = 4;
constexpr int MM  = BB * SS;       // 2048 token rows

// ---------------------------------------------------------------------------
// Embedding gather: x[row, :] = emb[ids[row], :]
// ---------------------------------------------------------------------------
__global__ __launch_bounds__(256) void embed_kernel(const int* __restrict__ ids,
                                                    const float* __restrict__ emb,
                                                    float* __restrict__ x) {
  int row = blockIdx.x;
  int tok = ids[row];
  const float* src = emb + (size_t)tok * DD;
  float* dst = x + (size_t)row * DD;
  for (int d = threadIdx.x; d < DD; d += 256) dst[d] = src[d];
}

// ---------------------------------------------------------------------------
// RMSNorm row -> bf16: out = x * rsqrt(mean(x^2)+eps) * g
// ---------------------------------------------------------------------------
__global__ __launch_bounds__(256) void rmsnorm_kernel(const float* __restrict__ x,
                                                      const float* __restrict__ g,
                                                      __bf16* __restrict__ out,
                                                      int D) {
  int row = blockIdx.x;
  const float* xr = x + (size_t)row * D;
  float s = 0.f;
  for (int d = threadIdx.x; d < D; d += 256) { float v = xr[d]; s += v * v; }
  __shared__ float red[256];
  red[threadIdx.x] = s;
  __syncthreads();
  for (int off = 128; off > 0; off >>= 1) {
    if (threadIdx.x < off) red[threadIdx.x] += red[threadIdx.x + off];
    __syncthreads();
  }
  float rinv = rsqrtf(red[0] / (float)D + 1e-5f);
  __bf16* o = out + (size_t)row * D;
  for (int d = threadIdx.x; d < D; d += 256) o[d] = (__bf16)(xr[d] * rinv * g[d]);
}

// ---------------------------------------------------------------------------
// Tiled WMMA GEMM:  Cout[M,N] = A[M,K](bf16) * W[N,K]^T(f32->bf16) (+ Cin)
// Block = 256 thr = 8 waves as 4(m) x 2(n); block tile 256x128, k-tile 32.
// Wave computes 64x64 = 16 WMMA accumulators. M%256==0, N%128==0, K%32==0.
// A tile (256 rows x 64 B, row stride 2K bytes) is staged global->LDS by the
// Tensor Data Mover: one tensor_load_to_lds per k-step issued by wave 0,
// completion via s_wait_tensorcnt. Fallback: per-lane async-to-LDS copies.
// ---------------------------------------------------------------------------
__global__ __launch_bounds__(256) void gemm_kernel(const __bf16* __restrict__ A,
                                                   const float* __restrict__ W,
                                                   const float* __restrict__ Cin,
                                                   float* __restrict__ Cout,
                                                   int M, int N, int K) {
  __shared__ __align__(16) __bf16 sA[256][32];   // 16 KB, row-major [m][k]
  __shared__ __align__(16) __bf16 sB[32][128];   //  8 KB, k-major  [k][n]

  const int tid  = threadIdx.x;
  const int lane = tid & 31;
  const int wave = tid >> 5;
  const int wm   = wave >> 1;      // 0..3 -> m offset wm*64
  const int wn   = wave & 1;       // 0..1 -> n offset wn*64
  const int m0   = blockIdx.x * 256;
  const int n0   = blockIdx.y * 128;
  const int half = lane >> 4;      // A/C half selector
  const int l15  = lane & 15;

  // LDS byte offset of sA (generic-pointer low 32 bits == LDS offset).
  const unsigned ldsA = (unsigned)(unsigned long long)(void*)&sA[0][0];

  const v8f vzero = {0.f, 0.f, 0.f, 0.f, 0.f, 0.f, 0.f, 0.f};
  v8f acc[4][4];
#pragma unroll
  for (int i = 0; i < 4; i++)
#pragma unroll
    for (int j = 0; j < 4; j++) acc[i][j] = vzero;

  for (int k0 = 0; k0 < K; k0 += 32) {
#ifdef USE_TDM
    // --- stage A tile via Tensor Data Mover (wave 0 issues, D# per ISA 8.3/8.4) ---
    if (wave == 0) {
      unsigned long long ga = (unsigned long long)(A + (size_t)m0 * K + k0);
      u32x4 g0;
      g0[0] = 1u;                                   // count=1, user mode
      g0[1] = ldsA;                                 // lds_addr
      g0[2] = (unsigned)(ga & 0xFFFFFFFFu);         // global_addr[31:0]
      g0[3] = (unsigned)((ga >> 32) & 0x1FFFFFFu)   // global_addr[56:32]
            | (2u << 30);                           // type = 2 ("image")
      i32x8 g1;
      g1[0] = (int)(1u << 16);                      // data_size=1 (2 bytes/elem)
      g1[1] = (int)((unsigned)(K & 0xFFFF) << 16);  // tensor_dim0[15:0]
      g1[2] = (int)(((unsigned)K >> 16) & 0xFFFFu)  // tensor_dim0[31:16]
            | (int)((unsigned)(M & 0xFFFF) << 16);  // tensor_dim1[15:0]
      g1[3] = (int)(((unsigned)M >> 16) & 0xFFFFu)  // tensor_dim1[31:16]
            | (int)(32u << 16);                     // tile_dim0 = 32 elems
      g1[4] = 256;                                  // tile_dim1 = 256 rows
      g1[5] = K;                                    // tensor_dim0_stride
      g1[6] = 0;
      g1[7] = 0;
      const i32x4 zz4 = {0, 0, 0, 0};
      const i32x8 zz8 = {0, 0, 0, 0, 0, 0, 0, 0};
      __builtin_amdgcn_tensor_load_to_lds(g0, g1, zz4, zz4, zz8, 0);
      __builtin_amdgcn_s_wait_tensorcnt(0);
    }
#else
    // --- fallback: per-lane async global->LDS copies (ASYNCcnt) ---
    {
      unsigned ldsRow = ldsA + (unsigned)tid * 64u;
      const __bf16* gsrc = A + (size_t)(m0 + tid) * K + k0;
      asm volatile("global_load_async_to_lds_b128 %0, %1, off"
                   :: "v"(ldsRow), "v"(gsrc) : "memory");
      asm volatile("global_load_async_to_lds_b128 %0, %1, off offset:16"
                   :: "v"(ldsRow), "v"(gsrc) : "memory");
      asm volatile("global_load_async_to_lds_b128 %0, %1, off offset:32"
                   :: "v"(ldsRow), "v"(gsrc) : "memory");
      asm volatile("global_load_async_to_lds_b128 %0, %1, off offset:48"
                   :: "v"(ldsRow), "v"(gsrc) : "memory");
      asm volatile("s_wait_asynccnt 0" ::: "memory");
    }
#endif
    // --- stage W tile, converting f32 -> bf16, transposed to [k][n] ---
    {
      int nl = tid >> 1;
      int kh = (tid & 1) * 16;
      const float4* src =
          reinterpret_cast<const float4*>(W + (size_t)(n0 + nl) * K + k0 + kh);
#pragma unroll
      for (int i = 0; i < 4; i++) {
        float4 f = src[i];
        sB[kh + i * 4 + 0][nl] = (__bf16)f.x;
        sB[kh + i * 4 + 1][nl] = (__bf16)f.y;
        sB[kh + i * 4 + 2][nl] = (__bf16)f.z;
        sB[kh + i * 4 + 3][nl] = (__bf16)f.w;
      }
      if (k0 + 32 < K) __builtin_prefetch(src + 8, 0, 1);
    }
    __syncthreads();

    // --- A fragments (ISA 7.12.2 16-bit A 16x32): lane row = l15, ---
    // --- chunks k = half*8+[0,8) and 16+half*8+[0,8)              ---
    v16bf afrag[4];
#pragma unroll
    for (int mt = 0; mt < 4; mt++) {
      const __bf16* src = &sA[wm * 64 + mt * 16 + l15][0];
#pragma unroll
      for (int j = 0; j < 8; j++) {
        afrag[mt][j]     = src[half * 8 + j];
        afrag[mt][8 + j] = src[16 + half * 8 + j];
      }
    }
    // --- B fragments: lane <-> n (l15), packed element j <-> k ---
    v16bf bfrag[4];
#pragma unroll
    for (int nt = 0; nt < 4; nt++) {
      const int nc = wn * 64 + nt * 16 + l15;
#pragma unroll
      for (int j = 0; j < 8; j++) {
        bfrag[nt][j]     = sB[half * 8 + j][nc];
        bfrag[nt][8 + j] = sB[16 + half * 8 + j][nc];
      }
    }
#pragma unroll
    for (int mt = 0; mt < 4; mt++)
#pragma unroll
      for (int nt = 0; nt < 4; nt++)
        acc[mt][nt] = WMMA_BF16(afrag[mt], bfrag[nt], acc[mt][nt]);
    __syncthreads();
  }

  // --- write out (C layout: m = r + 8*half, n = l15) ---
#pragma unroll
  for (int mt = 0; mt < 4; mt++) {
#pragma unroll
    for (int nt = 0; nt < 4; nt++) {
#pragma unroll
      for (int r = 0; r < 8; r++) {
        int row = m0 + wm * 64 + mt * 16 + r + 8 * half;
        int col = n0 + wn * 64 + nt * 16 + l15;
        float v = acc[mt][nt][r];
        if (Cin) v += Cin[(size_t)row * N + col];
        Cout[(size_t)row * N + col] = v;
      }
    }
  }
}

// ---------------------------------------------------------------------------
// RoPE + head layout: fp32 [B*S, D] (col = h*64+dh)  ->  bf16 [(b*H+h)*S+s][64]
// ---------------------------------------------------------------------------
__global__ __launch_bounds__(256) void rope_kernel(const float* __restrict__ in,
                                                   __bf16* __restrict__ out,
                                                   int do_rope) {
  int idx = blockIdx.x * 256 + threadIdx.x;        // over B*S*D/2 pairs
  if (idx >= MM * DD / 2) return;
  int j   = idx & 31;          // pair index within head dim (0..31)
  int h   = (idx >> 5) & 15;
  int row = idx >> 9;          // 0..2047
  int b = row >> 10, s = row & 1023;
  const float* src = in + (size_t)row * DD + h * DHD + 2 * j;
  float xe = src[0], xo = src[1];
  if (do_rope) {
    // ang = s * theta^(-2j/DH),  ln(10000) = 9.21034037...
    float ang = (float)s * __expf(-(2.f * (float)j / (float)DHD) * 9.210340371976184f);
    float c = __cosf(ang), sn = __sinf(ang);
    float re = c * xe - sn * xo;
    float ro = sn * xe + c * xo;
    xe = re; xo = ro;
  }
  __bf16* dst = out + ((size_t)(b * HHN + h) * SS + s) * DHD + 2 * j;
  dst[0] = (__bf16)xe;
  dst[1] = (__bf16)xo;
}

// ---------------------------------------------------------------------------
// Causal attention, one wave per (b,h, 16-row q tile).
// Scores row-resident in LDS (16 x 1024 f32 = 64 KB), two-pass softmax,
// QK^T and P*V via WMMA bf16. K^T fragments via global_load_b128 row loads.
// ---------------------------------------------------------------------------
__global__ __launch_bounds__(32) void attn_kernel(const __bf16* __restrict__ q,
                                                  const __bf16* __restrict__ k,
                                                  const __bf16* __restrict__ v,
                                                  __bf16* __restrict__ o) {
  __shared__ float sc[16][SS];                     // 64 KB
  const int bh   = blockIdx.x;                     // 0..B*H-1
  const int qt   = blockIdx.y;                     // 0..S/16-1
  const int lane = threadIdx.x;
  const int b = bh >> 4, h = bh & 15;
  const int q0   = qt * 16;
  const int half = lane >> 4;
  const int l15  = lane & 15;
  const __bf16* qp = q + (size_t)bh * SS * DHD;
  const __bf16* kp = k + (size_t)bh * SS * DHD;
  const __bf16* vp = v + (size_t)bh * SS * DHD;
  const v8f vzero = {0.f, 0.f, 0.f, 0.f, 0.f, 0.f, 0.f, 0.f};

  // Q A-fragments (K = 64 -> two 32-wide k-steps), held in registers.
  v16bf qa[2];
#pragma unroll
  for (int ks = 0; ks < 2; ks++) {
    const __bf16* src = qp + (size_t)(q0 + l15) * DHD + ks * 32;
#pragma unroll
    for (int j = 0; j < 8; j++) {
      qa[ks][j]     = src[half * 8 + j];
      qa[ks][8 + j] = src[16 + half * 8 + j];
    }
  }

  const float scale = 0.125f;                      // 1/sqrt(64)
  for (int kt = 0; kt <= qt; kt++) {
    v8f c = vzero;
#pragma unroll
    for (int ks = 0; ks < 2; ks++) {
      // B = K^T: two 16-byte row loads fill the v16bf B fragment.
      // Per-lane addr: key row = l15 (row stride 128 B), 16 B sub-chunk by
      // lane half; second 16-elem chunk via inst offset 32.
      v8bf t0, t1;
      const __bf16* base =
          kp + (size_t)(kt * 16 + l15) * DHD + ks * 32 + half * 8;
      asm volatile("global_load_b128 %0, %1, off"
                   : "=v"(t0) : "v"(base) : "memory");
      asm volatile("global_load_b128 %0, %1, off offset:32"
                   : "=v"(t1) : "v"(base) : "memory");
      asm volatile("s_wait_loadcnt 0x0" : "+v"(t0), "+v"(t1) :: "memory");
      v16bf bf = __builtin_shufflevector(t0, t1, 0, 1, 2, 3, 4, 5, 6, 7,
                                         8, 9, 10, 11, 12, 13, 14, 15);
      c = WMMA_BF16(qa[ks], bf, c);
    }
#pragma unroll
    for (int r = 0; r < 8; r++) {
      int m = r + 8 * half;
      int col = kt * 16 + l15;
      float vsc = c[r] * scale;
      if (col > q0 + m) vsc = -1e30f;              // causal mask
      sc[m][col] = vsc;
    }
  }
  __syncthreads();

  // Softmax over row length Lk; two lanes per row (strided), shfl_xor combine.
  const int Lk = (qt + 1) * 16;
  {
    int m = l15;
    float mx = -1e30f;
    for (int cdx = half; cdx < Lk; cdx += 2) mx = fmaxf(mx, sc[m][cdx]);
    mx = fmaxf(mx, __shfl_xor(mx, 16, 32));
    float sum = 0.f;
    for (int cdx = half; cdx < Lk; cdx += 2) {
      float e = __expf(sc[m][cdx] - mx);
      sc[m][cdx] = e;
      sum += e;
    }
    sum += __shfl_xor(sum, 16, 32);
    float inv = 1.f / sum;
    for (int cdx = half; cdx < Lk; cdx += 2) sc[m][cdx] *= inv;
  }
  __syncthreads();

  // O = P x V   (A = probs fp32->bf16 from LDS, B = V rows, 4 n-subtiles)
  v8f oc[4] = {vzero, vzero, vzero, vzero};
  for (int kk = 0; kk < Lk; kk += 32) {
    v16bf pa;
    const float* src = &sc[l15][kk];
#pragma unroll
    for (int j = 0; j < 8; j++) {
      pa[j]     = (__bf16)src[half * 8 + j];
      pa[8 + j] = (__bf16)src[16 + half * 8 + j];
    }
    const __bf16* vrow = vp + (size_t)kk * DHD;   // lane <-> n, element <-> k
#pragma unroll
    for (int nt = 0; nt < 4; nt++) {
      v16bf bf;
#pragma unroll
      for (int j = 0; j < 8; j++) {
        bf[j]     = vrow[(size_t)(half * 8 + j) * DHD + nt * 16 + l15];
        bf[8 + j] = vrow[(size_t)(16 + half * 8 + j) * DHD + nt * 16 + l15];
      }
      oc[nt] = WMMA_BF16(pa, bf, oc[nt]);
    }
  }

  // Write O back in [B*S, D] token-major layout (bf16, feeds Wo GEMM).
#pragma unroll
  for (int nt = 0; nt < 4; nt++) {
#pragma unroll
    for (int r = 0; r < 8; r++) {
      int m   = r + 8 * half;
      int row = b * SS + q0 + m;
      int col = h * DHD + nt * 16 + l15;
      o[(size_t)row * DD + col] = (__bf16)oc[nt][r];
    }
  }
}

// ---------------------------------------------------------------------------
// SwiGLU elementwise: out = bf16( u*sigmoid(u) * t )
// ---------------------------------------------------------------------------
__global__ __launch_bounds__(256) void swish_kernel(const float* __restrict__ u,
                                                    const float* __restrict__ t,
                                                    __bf16* __restrict__ out,
                                                    int n) {
  int i = blockIdx.x * 256 + threadIdx.x;
  if (i >= n) return;
  float uv = u[i];
  float g = uv / (1.f + __expf(-uv));
  out[i] = (__bf16)(g * t[i]);
}

// ---------------------------------------------------------------------------
// Host orchestration
// ---------------------------------------------------------------------------
extern "C" void kernel_launch(void* const* d_in, const int* in_sizes, int n_in,
                              void* d_out, int out_size, void* d_ws, size_t ws_size,
                              hipStream_t stream) {
  (void)in_sizes; (void)n_in; (void)out_size; (void)ws_size;
  const int*   ids = (const int*)  d_in[0];
  const float* emb = (const float*)d_in[1];
  const float* Wq  = (const float*)d_in[2];
  const float* Wk  = (const float*)d_in[3];
  const float* Wv  = (const float*)d_in[4];
  const float* Wo  = (const float*)d_in[5];
  const float* w1  = (const float*)d_in[6];
  const float* w2  = (const float*)d_in[7];
  const float* w3  = (const float*)d_in[8];
  const float* g1  = (const float*)d_in[9];
  const float* g2  = (const float*)d_in[10];
  const float* gf  = (const float*)d_in[11];
  const float* lmh = (const float*)d_in[12];
  float* out = (float*)d_out;

  // Workspace carve-out (~116 MB total).
  char* p = (char*)d_ws;
  auto carve = [&](size_t bytes) {
    void* r = (void*)p;
    p += (bytes + 255) & ~(size_t)255;
    return r;
  };
  float*  x_res = (float*) carve((size_t)MM * DD * 4);   // residual stream
  __bf16* h_bf  = (__bf16*)carve((size_t)MM * DD * 2);   // normalized acts
  float*  f1    = (float*) carve((size_t)MM * FF * 4);   // qf / u
  float*  f2    = (float*) carve((size_t)MM * FF * 4);   // kf / t
  float*  f3    = (float*) carve((size_t)MM * DD * 4);   // vf
  __bf16* q_bf  = (__bf16*)carve((size_t)MM * DD * 2);
  __bf16* k_bf  = (__bf16*)carve((size_t)MM * DD * 2);
  __bf16* v_bf  = (__bf16*)carve((size_t)MM * DD * 2);
  __bf16* o_bf  = (__bf16*)carve((size_t)MM * DD * 2);
  __bf16* g_bf  = (__bf16*)carve((size_t)MM * FF * 2);

  const dim3 blk(256);
  const dim3 gD(MM / 256, DD / 128);    // [2048 x 1024] outputs
  const dim3 gF(MM / 256, FF / 128);    // [2048 x 4096] outputs
  const dim3 gV(MM / 256, VV / 128);    // [2048 x 32000] outputs
  const int  ropeBlocks = (MM * DD / 2 + 255) / 256;

  embed_kernel<<<MM, blk, 0, stream>>>(ids, emb, x_res);

  for (int l = 0; l < LLN; l++) {
    const float* wq = Wq + (size_t)l * DD * DD;
    const float* wk = Wk + (size_t)l * DD * DD;
    const float* wv = Wv + (size_t)l * DD * DD;
    const float* wo = Wo + (size_t)l * DD * DD;
    const float* W1 = w1 + (size_t)l * FF * DD;
    const float* W2 = w2 + (size_t)l * DD * FF;
    const float* W3 = w3 + (size_t)l * FF * DD;

    rmsnorm_kernel<<<MM, blk, 0, stream>>>(x_res, g1 + (size_t)l * DD, h_bf, DD);
    gemm_kernel<<<gD, blk, 0, stream>>>(h_bf, wq, nullptr, f1, MM, DD, DD);
    gemm_kernel<<<gD, blk, 0, stream>>>(h_bf, wk, nullptr, f2, MM, DD, DD);
    gemm_kernel<<<gD, blk, 0, stream>>>(h_bf, wv, nullptr, f3, MM, DD, DD);
    rope_kernel<<<ropeBlocks, blk, 0, stream>>>(f1, q_bf, 1);
    rope_kernel<<<ropeBlocks, blk, 0, stream>>>(f2, k_bf, 1);
    rope_kernel<<<ropeBlocks, blk, 0, stream>>>(f3, v_bf, 0);

    attn_kernel<<<dim3(BB * HHN, SS / 16), dim3(32), 0, stream>>>(q_bf, k_bf, v_bf, o_bf);

    gemm_kernel<<<gD, blk, 0, stream>>>(o_bf, wo, x_res, x_res, MM, DD, DD);

    rmsnorm_kernel<<<MM, blk, 0, stream>>>(x_res, g2 + (size_t)l * DD, h_bf, DD);
    gemm_kernel<<<gF, blk, 0, stream>>>(h_bf, W1, nullptr, f1, MM, FF, DD);
    gemm_kernel<<<gF, blk, 0, stream>>>(h_bf, W3, nullptr, f2, MM, FF, DD);
    swish_kernel<<<(MM * FF + 255) / 256, blk, 0, stream>>>(f1, f2, g_bf, MM * FF);
    gemm_kernel<<<gD, blk, 0, stream>>>(g_bf, W2, x_res, x_res, MM, DD, FF);
  }

  rmsnorm_kernel<<<MM, blk, 0, stream>>>(x_res, gf, h_bf, DD);
  gemm_kernel<<<gV, blk, 0, stream>>>(h_bf, lmh, nullptr, out, MM, VV, DD);
}
